// LSTMpredictor_45320494907929
// MI455X (gfx1250) — matrix operand order repaired and checked by reading, not compiled, not run on hardware
//
#include <hip/hip_runtime.h>

// LSTM recurrence, B=1024, T=1024, H=51, input_size=1.
// Latency-bound serial scan -> keep all state on-chip, WMMA for the per-step
// [16x64]x[64x208] gate GEMM, weights pinned in VGPRs, h in LDS (f16), c in regs.
// v4: raw v_exp_f32 via __builtin_amdgcn_exp2f (no OCML range-fixup sequence);
//     bias in WMMA C operand; packed FMA/mul gate epilogue; vectorized fc.

typedef _Float16 v16h __attribute__((ext_vector_type(16)));
typedef _Float16 v8h  __attribute__((ext_vector_type(8)));
typedef float    v8f  __attribute__((ext_vector_type(8)));

#define HID 51
#define G4 204          // 4*HID
#define GS 208          // gates LDS row stride (padded)
#define KP 64           // K padded to 2x WMMA K=32
#define HS 52           // f32 h row stride: 52*m mod 64 distinct for m=0..15
#define NTILES 13       // ceil(204/16) N-tiles of the gate matrix
#define MT 16           // batch tile per workgroup (WMMA M)
#define NTHREADS (NTILES * 32)   // 416 threads = 13 waves (wave32)
#define ELEMS (MT * HID)         // 816 (m,j) state elements per block
#define LOG2E 1.4426950408889634f

__device__ __forceinline__ float rcp_fast(float x)  { return __builtin_amdgcn_rcpf(x); }
__device__ __forceinline__ float exp2_fast(float x) { return __builtin_amdgcn_exp2f(x); }
// tanh(x) = 2/(1+exp2(x * -2*log2e)) - 1  (one mul + v_exp + v_rcp + fma)
__device__ __forceinline__ float tanh_fast(float x) {
    return 2.0f * rcp_fast(1.0f + exp2_fast(x * (-2.0f * LOG2E))) - 1.0f;
}

__global__ void __launch_bounds__(NTHREADS)
lstm_wmma_kernel(const float* __restrict__ x,      // [B,T]
                 const float* __restrict__ W_ih,   // [4H,1]
                 const float* __restrict__ W_hh,   // [4H,H]
                 const float* __restrict__ b_ih,   // [4H]
                 const float* __restrict__ b_hh,   // [4H]
                 const float* __restrict__ fc_w,   // [1,H]
                 const float* __restrict__ fc_b,   // [1]
                 float* __restrict__ out,          // [B,T]
                 int T)
{
    const int tid  = threadIdx.x;
    const int lane = tid & 31;
    const int wid  = tid >> 5;              // N-tile index 0..12
    const int b0   = blockIdx.x * MT;       // batch base row

    __shared__ __align__(16) _Float16 h_lds[MT * KP];   // 2 KB f16 A-matrix, K padded
    __shared__ __align__(16) float h32_lds[MT * HS];    // f32 h copy for fc (stride 52)
    __shared__ float gates_lds[MT * GS];                // 13 KB activated gates
    __shared__ float x_lds[MT];
    __shared__ __align__(16) float fcw_lds[HS];         // fc_w padded with zero

    // ---------------- one-time init ----------------
    for (int i = tid; i < MT * KP; i += NTHREADS) h_lds[i] = (_Float16)0.0f;
    for (int i = tid; i < MT * HS; i += NTHREADS) h32_lds[i] = 0.0f;
    if (tid < HS)  fcw_lds[tid] = (tid < HID) ? fc_w[tid] : 0.0f;
    if (tid < MT)  x_lds[tid]   = x[(size_t)(b0 + tid) * T];   // x_t for t=0

    // B-fragment (W_hh^T tile, column n = 16*wid + lane%16), pinned in VGPRs.
    // 16-bit B 32x16 layout: lane holds column n; lanes 0-15 K=0..7,16..23, lanes
    // 16-31 K=8..15,24..31 (half index i -> K = kb2 + i (+8 for i>=8)); hi fragment is K+32.
    const int  ncol    = wid * 16 + (lane & 15);
    const bool ncol_ok = (ncol < G4);
    const int  kb2     = (lane >> 4) * 8;
    v16h bfrag_lo, bfrag_hi;
#pragma unroll
    for (int i = 0; i < 16; ++i) {
        int k0 = kb2 + i + (i >= 8 ? 8 : 0);
        int k1 = k0 + 32;
        float v0 = (ncol_ok && k0 < HID) ? W_hh[ncol * HID + k0] : 0.0f;
        float v1 = (ncol_ok && k1 < HID) ? W_hh[ncol * HID + k1] : 0.0f;
        bfrag_lo[i] = (_Float16)v0;
        bfrag_hi[i] = (_Float16)v1;
    }
    const float wih_n  = ncol_ok ? W_ih[ncol] : 0.0f;
    const float bias_n = ncol_ok ? (b_ih[ncol] + b_hh[ncol]) : 0.0f;
    // Bias rides through the matrix unit as the C operand of the first WMMA.
    v8f bias_acc;
#pragma unroll
    for (int r = 0; r < 8; ++r) bias_acc[r] = bias_n;

    // PyTorch gate order i,f,g,o: 'g' (n in [2H,3H)) uses tanh, rest sigmoid.
    // Branch-free: act(g) = fma(gsc, rcp(1 + exp2(g*gk)), -gofs)
    //   sigmoid: gsc=1, gofs=0, gk=-log2e ; tanh: gsc=2, gofs=1, gk=-2*log2e
    const bool  tanh_gate = (ncol >= 2 * HID) && (ncol < 3 * HID);
    const float gsc  = tanh_gate ? 2.0f : 1.0f;
    const float gofs = tanh_gate ? 1.0f : 0.0f;
    const float gk   = -gsc * LOG2E;
    const float fcb  = fc_b[0];

    // Per-thread c-state ownership: elements e0=tid, e1=tid+416 of 816.
    const int e0 = tid, e1 = tid + NTHREADS;
    const int m0 = e0 / HID, j0 = e0 - m0 * HID;
    const int m1 = e1 / HID, j1 = e1 - m1 * HID;
    float c0 = 0.0f, c1 = 0.0f;

    // A-fragment LDS base (16-bit A 16x32 layout): lane holds row m=lane%16;
    // lanes 0-15 K base 0, lanes 16-31 K base 8; halves 8..15 at K+16.
    const int arow = lane & 15;
    const int akb  = (lane >> 4) * 8;
    const _Float16* hp = &h_lds[arow * KP + akb];

    // fc projection split: lane = (row m, half); half0 j=[0,32), half1 j=[32,52)
    const int fm   = lane & 15;
    const int fh   = lane >> 4;
    const float4* fhp = (const float4*)&h32_lds[fm * HS + fh * 32];
    const float4* fwp = (const float4*)&fcw_lds[fh * 32];
    const int fn4 = fh ? 5 : 8;

    __syncthreads();

    // ---------------- serial recurrence ----------------
    for (int t = 0; t < T; ++t) {
        // phase 1: gates = h @ W_hh^T + bias  (2 chained WMMAs, K=64)
        union { v16h v; v8h p[2]; } a_lo, a_hi;
        a_lo.p[0] = *(const v8h*)(hp);        // K akb   .. akb+7   (16B aligned)
        a_lo.p[1] = *(const v8h*)(hp + 16);   // K akb+16.. akb+23
        a_hi.p[0] = *(const v8h*)(hp + 32);   // K akb+32.. akb+39
        a_hi.p[1] = *(const v8h*)(hp + 48);   // K akb+48.. akb+55
        v8f acc = __builtin_amdgcn_wmma_f32_16x16x32_f16(false, a_lo.v, false, bfrag_lo,
                                                         (short)0, bias_acc, false, false);
        acc = __builtin_amdgcn_wmma_f32_16x16x32_f16(false, a_hi.v, false, bfrag_hi,
                                                     (short)0, acc, false, false);
        if (ncol_ok) {
            const int mofs = (lane >> 4) * 8;   // C layout: VGPR r -> M = r (+8 hi lanes)
#pragma unroll
            for (int r = 0; r < 8; ++r) {
                int m = r + mofs;
                float g = __builtin_fmaf(x_lds[m], wih_n, acc[r]);  // + x_t*W_ih
                float a = __builtin_fmaf(gsc, rcp_fast(1.0f + exp2_fast(g * gk)), -gofs);
                gates_lds[m * GS + ncol] = a;
            }
        }
        __syncthreads();   // gates visible; gates buffer free from prev step

        // phase 2: elementwise c/h update (c in registers, h -> LDS f16 + f32)
        {
            float iv = gates_lds[m0 * GS + j0];
            float fv = gates_lds[m0 * GS + j0 + HID];
            float gv = gates_lds[m0 * GS + j0 + 2 * HID];
            float ov = gates_lds[m0 * GS + j0 + 3 * HID];
            c0 = fv * c0 + iv * gv;
            float hv = ov * tanh_fast(c0);
            h_lds[m0 * KP + j0]  = (_Float16)hv;
            h32_lds[m0 * HS + j0] = hv;
        }
        if (e1 < ELEMS) {
            float iv = gates_lds[m1 * GS + j1];
            float fv = gates_lds[m1 * GS + j1 + HID];
            float gv = gates_lds[m1 * GS + j1 + 2 * HID];
            float ov = gates_lds[m1 * GS + j1 + 3 * HID];
            c1 = fv * c1 + iv * gv;
            float hv = ov * tanh_fast(c1);
            h_lds[m1 * KP + j1]  = (_Float16)hv;
            h32_lds[m1 * HS + j1] = hv;
        }
        if (tid < MT && (t + 1) < T)   // prefetch next x_t
            x_lds[tid] = x[(size_t)(b0 + tid) * T + (t + 1)];
        __syncthreads();   // new h visible

        // output projection: out[b][t] = h . fc_w + fc_b, wave 0, 32 lanes
        // (runs while other waves proceed into next step's phase 1)
        if (wid == 0) {
            float s = fh ? 0.0f : fcb;
#pragma unroll
            for (int i = 0; i < 8; ++i) {
                if (i < fn4) {
                    float4 hv = fhp[i];
                    float4 wv = fwp[i];
                    s += hv.x * wv.x + hv.y * wv.y + hv.z * wv.z + hv.w * wv.w;
                }
            }
            s += __shfl_xor(s, 16, 32);   // combine the two half-row sums
            if (fh == 0)
                out[(size_t)(b0 + fm) * T + t] = s;
        }
    }
}

extern "C" void kernel_launch(void* const* d_in, const int* in_sizes, int n_in,
                              void* d_out, int out_size, void* d_ws, size_t ws_size,
                              hipStream_t stream) {
    const float* x    = (const float*)d_in[0];
    const float* W_ih = (const float*)d_in[1];
    const float* W_hh = (const float*)d_in[2];
    const float* b_ih = (const float*)d_in[3];
    const float* b_hh = (const float*)d_in[4];
    const float* fc_w = (const float*)d_in[5];
    const float* fc_b = (const float*)d_in[6];
    float* out = (float*)d_out;

    const int T = 1024;
    const int B = in_sizes[0] / T;      // 1024
    dim3 grid(B / MT);                  // 64 independent recurrences
    lstm_wmma_kernel<<<grid, NTHREADS, 0, stream>>>(x, W_ih, W_hh, b_ih, b_hh,
                                                    fc_w, fc_b, out, T);
}
